// PoolingCapsuleLayer_41008347742651
// MI455X (gfx1250) — matrix operand order repaired and checked by reading, not compiled, not run on hardware
//
#include <hip/hip_runtime.h>
#include <hip/hip_bf16.h>
#include <math.h>

// ---------------------------------------------------------------------------
// Problem constants (from reference): B=8, H=W=32, Cin=32, Cout=64, pool=2x2
// ---------------------------------------------------------------------------
#define NB     8
#define HH     32
#define WW     32
#define CIN    32
#define COUT   64
#define HP     16
#define WP     16
#define NSITES (NB * HP * WP)     // 2048
#define NROWS  (NSITES * 4)       // 8192 rows of tpos/theta
#define TCOLS  (COUT * CIN)       // 2048 theta columns
#define CT_CHUNKS 8               // column-tile chunks per row tile (128/8=16 cts each)

typedef __attribute__((ext_vector_type(16))) _Float16 v16h;
typedef __attribute__((ext_vector_type(8)))  _Float16 v8h;
typedef __attribute__((ext_vector_type(8)))  float    v8f;

// ---------------------------------------------------------------------------
// K0: f32 -> f16 weight conversion (tg1_w 32x64, tg2_w 2048x32)
// ---------------------------------------------------------------------------
__global__ __launch_bounds__(256) void convert_f16_kernel(
    const float* __restrict__ src, _Float16* __restrict__ dst, int n) {
  int i = blockIdx.x * blockDim.x + threadIdx.x;
  if (i < n) dst[i] = (_Float16)src[i];
}

// ---------------------------------------------------------------------------
// K1: per-site channel-mean pose + tpos generation.
// One wave per site, lane == cin.  tposH layout: [8192][64] f16 row-major,
// row = site*4 + p, feature = cin*2 + {x,y}.
// ---------------------------------------------------------------------------
__global__ __launch_bounds__(128) void tpos_kernel(
    const float* __restrict__ pose, _Float16* __restrict__ tposH) {
  const int wave = threadIdx.x >> 5;
  const int lane = threadIdx.x & 31;
  const int site = blockIdx.x * 4 + wave;         // grid = 512 blocks
  const int b  = site >> 8;
  const int rm = site & 255;
  const int hp = rm >> 4, wp = rm & 15;
  const int cin = lane;

  float mx = 0.f, my = 0.f;
  #pragma unroll
  for (int p = 0; p < 4; ++p) {
    const int h = hp * 2 + (p >> 1);
    const int w = wp * 2 + (p & 1);
    const float* pp = pose + ((((size_t)b * HH + h) * WW + w) * CIN + cin) * 2;
    mx += pp[0];
    my += pp[1];
  }
  // normalize with zero-norm guard: m / (|m| + (1 - sign(|m|)))
  const float nrm = sqrtf(mx * mx + my * my);
  const float den = nrm + (nrm == 0.f ? 1.f : 0.f);
  const float c = mx / den, s = my / den;

  // rot = [[c, s], [-s, c]] applied to pool offsets (+-0.5, +-0.5)
  #pragma unroll
  for (int p = 0; p < 4; ++p) {
    const float oh = (p >> 1) ? 0.5f : -0.5f;
    const float ow = (p & 1) ? 0.5f : -0.5f;
    const float tx = c * oh + s * ow;
    const float ty = -s * oh + c * ow;
    _Float16* o = tposH + (size_t)(site * 4 + p) * (2 * CIN) + 2 * cin;
    o[0] = (_Float16)tx;
    o[1] = (_Float16)ty;
  }
}

// ---------------------------------------------------------------------------
// K2: theta = (tpos @ tg1^T) @ tg2^T via v_wmma_f32_16x16x32_f16.
// Block = 256 (8 waves).  Work unit = (row-tile, ct-chunk): 512 row tiles x
// 8 chunks = 4096 waves (8x the parallelism of tiling rows only; stage-1
// hidden is recomputed per chunk, +21% WMMAs, negligible vs latency hiding).
// Stage 1: hidden(16x32) in 4 WMMAs (K=64 -> 2 k-steps, 2 n-tiles), bounced
// through per-wave LDS to re-stripe D-layout -> A-layout.
// Stage 2: 16 col tiles x 1 WMMA each (K=32), D stored f32 to workspace,
// next B fragment prefetched (global_prefetch_b8).
// ---------------------------------------------------------------------------
__global__ __launch_bounds__(256) void gemm_theta_kernel(
    const _Float16* __restrict__ tposH,   // [8192][64]
    const _Float16* __restrict__ tg1H,    // [32][64]
    const _Float16* __restrict__ tg2H,    // [2048][32]
    float* __restrict__ theta)            // [8192][2048]
{
  __shared__ _Float16 hid[8][16 * 32];    // 8 KB, one 16x32 tile per wave
  const int wave = threadIdx.x >> 5;
  const int lane = threadIdx.x & 31;
  const int gw    = blockIdx.x * 8 + wave;    // 0..4095
  const int rt    = gw >> 3;                  // row tile 0..511
  const int chunk = gw & (CT_CHUNKS - 1);     // ct chunk 0..7
  const int row0 = rt * 16;
  const int n = lane & 15;                // row (A) / col (B,D) within tile
  const int h = lane >> 4;                // lane-half select
  _Float16* hlds = hid[wave];

  // ---- stage 1: hidden = tpos @ tg1^T ----
  #pragma unroll
  for (int nt = 0; nt < 2; ++nt) {
    v8f acc = {};
    #pragma unroll
    for (int ks = 0; ks < 2; ++ks) {
      // A frag: M = n, K chunks at 32*ks + 8h and 32*ks + 16 + 8h
      const _Float16* arow = tposH + (size_t)(row0 + n) * 64 + 32 * ks + 8 * h;
      v8h alo = *(const v8h*)arow;
      v8h ahi = *(const v8h*)(arow + 16);
      v16h afrag = __builtin_shufflevector(alo, ahi,
          0,1,2,3,4,5,6,7,8,9,10,11,12,13,14,15);
      // B frag: B[k, nn] = tg1[nt*16+n][k], K chunks at 32*ks + 8h and 32*ks + 16 + 8h
      const _Float16* brow = tg1H + (size_t)(nt * 16 + n) * 64 + 32 * ks + 8 * h;
      v8h blo = *(const v8h*)brow;
      v8h bhi = *(const v8h*)(brow + 16);
      v16h bfrag = __builtin_shufflevector(blo, bhi,
          0,1,2,3,4,5,6,7,8,9,10,11,12,13,14,15);
      acc = __builtin_amdgcn_wmma_f32_16x16x32_f16(
          false, afrag, false, bfrag, (short)0, acc, false, false);
    }
    // D element j -> hidden row M = j + 8h, col nt*16 + n  (store as f16)
    #pragma unroll
    for (int j = 0; j < 8; ++j)
      hlds[(j + 8 * h) * 32 + nt * 16 + n] = (_Float16)acc[j];
  }
  // same-wave DS ops are in order; no cross-wave sharing -> no barrier needed

  // ---- re-stripe hidden into stage-2 A fragment ----
  v8h a2lo = *(const v8h*)&hlds[n * 32 + 8 * h];        // j = 8h..8h+7
  v8h a2hi = *(const v8h*)&hlds[n * 32 + 16 + 8 * h];   // j = 16+8h..
  v16h a2 = __builtin_shufflevector(a2lo, a2hi,
      0,1,2,3,4,5,6,7,8,9,10,11,12,13,14,15);

  // ---- stage 2: theta = hidden @ tg2^T, this wave's 16 column tiles ----
  const int ct0 = chunk * (128 / CT_CHUNKS);
  const int ct1 = ct0 + (128 / CT_CHUNKS);
  for (int ct = ct0; ct < ct1; ++ct) {
    // B frag: B[j, c] = tg2[ct*16+n][j], j = 8h.. and 16+8h.. (16B chunks)
    const _Float16* b2row = tg2H + (size_t)(ct * 16 + n) * 32 + 8 * h;
    // prefetch next iteration's B fragment row into cache
    if (ct + 1 < ct1)
      __builtin_prefetch(b2row + 16 * 32, 0, 3);
    v8h b2lo = *(const v8h*)b2row;
    v8h b2hi = *(const v8h*)(b2row + 16);
    v16h b2 = __builtin_shufflevector(b2lo, b2hi,
        0,1,2,3,4,5,6,7,8,9,10,11,12,13,14,15);
    v8f zero = {};
    v8f d = __builtin_amdgcn_wmma_f32_16x16x32_f16(
        false, a2, false, b2, (short)0, zero, false, false);
    // D element j -> theta[row0 + j + 8h][ct*16 + n]
    float* out = theta + (size_t)(row0 + 8 * h) * TCOLS + ct * 16 + n;
    #pragma unroll
    for (int j = 0; j < 8; ++j)
      out[(size_t)j * TCOLS] = d[j];
  }
}

// ---------------------------------------------------------------------------
// K3: routing.  One wave per (site, cout) pair; lane == cin, each lane owns
// the 4 pool positions of its channel (128 votes / 32 lanes).
// 4 mean-normalize rounds (init + NUM_ITERS=3) with wave32 butterflies.
// ---------------------------------------------------------------------------
__global__ __launch_bounds__(256) void routing_kernel(
    const float* __restrict__ a, const float* __restrict__ pose,
    const float* __restrict__ theta, const float* __restrict__ alpha,
    const float* __restrict__ beta,
    float* __restrict__ agreement, float* __restrict__ pose_out) {
  const int wave = threadIdx.x >> 5;
  const int lane = threadIdx.x & 31;
  const int pair = blockIdx.x * 8 + wave;   // 0 .. 131071
  const int site = pair >> 6;
  const int cout = pair & 63;
  const int b  = site >> 8;
  const int rm = site & 255;
  const int hp = rm >> 4, wp = rm & 15;
  const int cin = lane;

  float vx[4], vy[4], aw[4], wgt[4];
  #pragma unroll
  for (int p = 0; p < 4; ++p) {
    const int h = hp * 2 + (p >> 1);
    const int w = wp * 2 + (p & 1);
    const size_t base = (((size_t)b * HH + h) * WW + w) * CIN + cin;
    const float px = pose[base * 2], py = pose[base * 2 + 1];
    aw[p] = a[base];
    const float th = theta[(size_t)(site * 4 + p) * TCOLS + cout * CIN + cin];
    const float sn = __sinf(th);
    const float cs = __cosf(th);
    vx[p] = px * cs - py * sn;   // R(pose)·R(theta) first column
    vy[p] = py * cs + px * sn;
    wgt[p] = aw[p];              // initial weight = broadcast activation
  }

  float pox = 0.f, poy = 0.f;
  #pragma unroll
  for (int it = 0; it <= 3; ++it) {
    if (it > 0) {
      #pragma unroll
      for (int p = 0; p < 4; ++p) {
        const float dot = pox * vx[p] + poy * vy[p];
        wgt[p] = (1.f + dot) * 0.5f * aw[p];   // (1 - dist) * a_e
      }
    }
    float mx = 0.f, my = 0.f;
    #pragma unroll
    for (int p = 0; p < 4; ++p) { mx += wgt[p] * vx[p]; my += wgt[p] * vy[p]; }
    #pragma unroll
    for (int m = 16; m >= 1; m >>= 1) {
      mx += __shfl_xor(mx, m, 32);
      my += __shfl_xor(my, m, 32);
    }
    const float nn = sqrtf(mx * mx + my * my);
    pox = mx / (nn + (mx == 0.f ? 1.f : 0.f));  // elementwise zero guard
    poy = my / (nn + (my == 0.f ? 1.f : 0.f));
  }

  float nd = 0.f, ws = 0.f;
  #pragma unroll
  for (int p = 0; p < 4; ++p) {
    const float dot = pox * vx[p] + poy * vy[p];
    nd += (1.f + dot) * 0.5f * wgt[p];
    ws += (wgt[p] != 0.f) ? 1.f : 0.f;          // sign(|w|)
  }
  #pragma unroll
  for (int m = 16; m >= 1; m >>= 1) {
    nd += __shfl_xor(nd, m, 32);
    ws += __shfl_xor(ws, m, 32);
  }
  if (lane == 0) {
    const float msk = (ws != 0.f) ? 1.f : 0.f;
    const float ndn = nd / (ws + (1.f - msk)) * msk;
    const float z = alpha[0] * ndn + (beta[0] - 1.f);
    const float ag = (1.f / (1.f + __expf(-z))) * msk;
    agreement[(size_t)site * COUT + cout] = ag;
    pose_out[((size_t)site * COUT + cout) * 2]     = pox;
    pose_out[((size_t)site * COUT + cout) * 2 + 1] = poy;
  }
}

// ---------------------------------------------------------------------------
// K4: feature max-pool + linear.  64 threads per site (thread == cout),
// first 32 threads also compute/write xp through LDS.
// ---------------------------------------------------------------------------
__global__ __launch_bounds__(64) void pool_linear_kernel(
    const float* __restrict__ x, const float* __restrict__ lin_w,
    const float* __restrict__ lin_b, float* __restrict__ x_out,
    float* __restrict__ xp_out) {
  __shared__ float xps[CIN];
  const int site = blockIdx.x;
  const int b  = site >> 8;
  const int rm = site & 255;
  const int hp = rm >> 4, wp = rm & 15;
  const int t = threadIdx.x;
  if (t < CIN) {
    float m = -INFINITY;
    #pragma unroll
    for (int p = 0; p < 4; ++p) {
      const int h = hp * 2 + (p >> 1);
      const int w = wp * 2 + (p & 1);
      m = fmaxf(m, x[(((size_t)b * HH + h) * WW + w) * CIN + t]);
    }
    xps[t] = m;
    xp_out[(size_t)site * CIN + t] = m;
  }
  __syncthreads();
  float acc = lin_b[t];
  #pragma unroll
  for (int cin = 0; cin < CIN; ++cin)
    acc += xps[cin] * lin_w[t * CIN + cin];
  x_out[(size_t)site * COUT + t] = acc;
}

// ---------------------------------------------------------------------------
// Host-side launch
// ---------------------------------------------------------------------------
extern "C" void kernel_launch(void* const* d_in, const int* in_sizes, int n_in,
                              void* d_out, int out_size, void* d_ws, size_t ws_size,
                              hipStream_t stream) {
  const float* x     = (const float*)d_in[0];
  const float* a     = (const float*)d_in[1];
  const float* pose  = (const float*)d_in[2];
  const float* alpha = (const float*)d_in[3];
  const float* beta  = (const float*)d_in[4];
  const float* lin_w = (const float*)d_in[5];
  const float* lin_b = (const float*)d_in[6];
  const float* tg1_w = (const float*)d_in[7];
  const float* tg2_w = (const float*)d_in[8];

  float* out       = (float*)d_out;
  float* x_out     = out;                               // 2048*64
  float* agreement = out + NSITES * COUT;               // 2048*64
  float* pose_o    = out + 2 * NSITES * COUT;           // 2048*64*2
  float* xp        = out + 4 * NSITES * COUT;           // 2048*32

  char* ws = (char*)d_ws;
  _Float16* tposH = (_Float16*)ws;                             // 1 MiB
  _Float16* tg1H  = (_Float16*)(ws + (1u << 20));              // 4 KiB
  _Float16* tg2H  = (_Float16*)(ws + (1u << 20) + 4096);       // 128 KiB
  float*    theta = (float*)(ws + (1u << 20) + 4096 + (128u << 10)); // 64 MiB

  // K0: weight conversion
  convert_f16_kernel<<<(CIN * 2 * CIN + 255) / 256, 256, 0, stream>>>(
      tg1_w, tg1H, CIN * 2 * CIN);                   // 32 x 64
  convert_f16_kernel<<<(TCOLS * CIN + 255) / 256, 256, 0, stream>>>(
      tg2_w, tg2H, TCOLS * CIN);                     // 2048 x 32

  // K1: tpos (one wave per site)
  tpos_kernel<<<NSITES / 4, 128, 0, stream>>>(pose, tposH);

  // K2: WMMA GEMM -> theta ((row-tile, ct-chunk) per wave, 8 waves/block)
  gemm_theta_kernel<<<(NROWS / 16) * CT_CHUNKS / 8, 256, 0, stream>>>(
      tposH, tg1H, tg2H, theta);

  // K3: routing (one wave per (site, cout))
  routing_kernel<<<(NSITES * COUT) / 8, 256, 0, stream>>>(
      a, pose, theta, alpha, beta, agreement, pose_o);

  // K4: max-pool + linear
  pool_linear_kernel<<<NSITES, 64, 0, stream>>>(x, lin_w, lin_b, x_out, xp);
}
